// GNNBlock_14663018349022
// MI455X (gfx1250) — matrix operand order repaired and checked by reading, not compiled, not run on hardware
//
#include <hip/hip_runtime.h>

typedef unsigned short ushort_t;
typedef __attribute__((ext_vector_type(16))) __bf16 v16bf;
typedef __attribute__((ext_vector_type(8)))  float  v8f;

#define NB   8192
#define NN   21
#define NP   32
#define DD   256
#define HH   8
#define HDIM 32
#define XSZ  ((size_t)NB * NN * DD)          // x output elems
#define ASZ  ((size_t)NB * HH * NN * NN)     // one attn tensor elems
#define WSZ  (2 * DD * DD)                   // one [L,D,D] weight elems

__device__ __constant__ unsigned int ADJ_MASK[NN] = {
    0x22223u, 0x7u, 0xEu, 0x1Cu, 0x18u,
    0x61u, 0xE0u, 0x1C0u, 0x180u,
    0x601u, 0xE00u, 0x1C00u, 0x1800u,
    0x6001u, 0xE000u, 0x1C000u, 0x18000u,
    0x60001u, 0xE0000u, 0x1C0000u, 0x180000u};

struct B256 { int4 a, b; };   // 32 bytes == 16 bf16

__device__ __forceinline__ ushort_t f2bf(float f) {
    unsigned int u = __float_as_uint(f);
    u = (u + 0x7FFFu + ((u >> 16) & 1u)) >> 16;
    return (ushort_t)u;
}

// A-operand (16x32 bf16): lane l<16 -> row r=l, K {0..7,16..23}; l>=16 -> row l-16, K {8..15,24..31}
__device__ __forceinline__ v16bf load_tileA(const ushort_t* buf, int stride, int r0, int c0) {
    int l = threadIdx.x & 31;
    int row = r0 + (l & 15);
    int off = (l >> 4) * 8;
    B256 t;
    t.a = *(const int4*)(buf + row * stride + c0 + off);
    t.b = *(const int4*)(buf + row * stride + c0 + 16 + off);
    return __builtin_bit_cast(v16bf, t);
}

// B-operand (32x16 bf16) from column-major-per-column storage (row = output column, contiguous K):
// lane l -> column l&15, K {0..7,16..23} for l<16; {8..15,24..31} for l>=16
__device__ __forceinline__ v16bf load_tileB(const ushort_t* buf, int stride, int c0, int k0) {
    int l = threadIdx.x & 31;
    int row = c0 + (l & 15);
    int base = k0 + ((l >> 4) ? 8 : 0);
    B256 t;
    t.a = *(const int4*)(buf + row * stride + base);
    t.b = *(const int4*)(buf + row * stride + base + 16);
    return __builtin_bit_cast(v16bf, t);
}

__device__ __forceinline__ v8f wmma_bf16(v16bf a, v16bf b, v8f c) {
    return __builtin_amdgcn_wmma_f32_16x16x32_bf16(false, a, false, b, (short)0, c, false, false);
}

// D tile (16x16 f32): lane l -> col n0+(l&15), rows m0+{0..7}+((l>=16)?8:0)
__device__ __forceinline__ void storeD_bf16(ushort_t* buf, int stride, int m0, int n0, v8f acc,
                                            const float* bias) {
    int l = threadIdx.x & 31;
    int col = n0 + (l & 15);
    int mb = m0 + ((l >> 4) ? 8 : 0);
    float bv = bias ? bias[col] : 0.0f;
#pragma unroll
    for (int r = 0; r < 8; ++r) buf[(mb + r) * stride + col] = f2bf(acc[r] + bv);
}

__device__ __forceinline__ void storeD_f32(float* buf, int stride, int m0, int n0, v8f acc) {
    int l = threadIdx.x & 31;
    int col = n0 + (l & 15);
    int mb = m0 + ((l >> 4) ? 8 : 0);
#pragma unroll
    for (int r = 0; r < 8; ++r) buf[(mb + r) * stride + col] = acc[r];
}

// store D transposed as bf16 into vT[256][32]: one packed b128 per lane
__device__ __forceinline__ void storeD_vT(ushort_t* vt, int m0, int c0, v8f acc, const float* bias) {
    int l = threadIdx.x & 31;
    int col = c0 + (l & 15);                 // feature index
    int mb = m0 + ((l >> 4) ? 8 : 0);        // token index base
    float bv = bias[col];
    union { ushort_t u[8]; int4 v; } p;
#pragma unroll
    for (int r = 0; r < 8; ++r) p.u[r] = f2bf(acc[r] + bv);
    *(int4*)(vt + col * 32 + mb) = p.v;
}

__global__ void prep_weights_kernel(const float* wq, const float* wk, const float* wv,
                                    const float* wo, const float* glw, ushort_t* ws) {
    int idx = blockIdx.x * 256 + threadIdx.x;
    if (idx < 4 * WSZ) {
        int which = idx / WSZ, off = idx % WSZ;
        const float* src = (which == 0) ? wq : (which == 1) ? wk : (which == 2) ? wv : wo;
        ws[idx] = f2bf(src[off]);
    } else if (idx < 4 * WSZ + DD * DD) {
        int g = idx - 4 * WSZ;
        int e = g >> 8, d = g & 255;
        ws[idx] = f2bf(glw[d * DD + e]);   // glwT[e][d] = gl_w[d][e]
    }
}

__launch_bounds__(256)
__global__ void gnn_block_kernel(const float* __restrict__ x_in,
                                 const float* __restrict__ bone,
                                 const float* __restrict__ ln_g,
                                 const float* __restrict__ ln_b,
                                 const float* __restrict__ bq,
                                 const float* __restrict__ bk,
                                 const float* __restrict__ bv,
                                 const float* __restrict__ bo,
                                 const ushort_t* __restrict__ wbf,
                                 float* __restrict__ out) {
    __shared__ float    s_x[NN * DD];          // 21.0 KB  (residual, f32)
    __shared__ ushort_t s_f[NP * DD];          // 16 KB    (LN(+bone) bf16, A-layout rows)
    __shared__ ushort_t s_q[NP * DD];          // 16 KB    (q / xt / finalA)
    __shared__ ushort_t s_k[NP * DD];          // 16 KB
    __shared__ ushort_t s_vT[DD * 32];         // 16 KB    (v transposed, per-feature rows)
    __shared__ float    s_scores[HH * 32 * 32];// 32 KB    (scores f32; reused as bf16 "out")
    __shared__ float    s_dadj[32 * 32];       // 4 KB

    const int b = blockIdx.x;
    const int tid = threadIdx.x;
    const int wave = tid >> 5;
    const int lane = tid & 31;
    const float scale = 0.17677669529663687f;  // 1/sqrt(32)

    const ushort_t* wq_bf = wbf;
    const ushort_t* wk_bf = wbf + WSZ;
    const ushort_t* wv_bf = wbf + 2 * WSZ;
    const ushort_t* wo_bf = wbf + 3 * WSZ;
    const ushort_t* glwT  = wbf + 4 * WSZ;

    // load x[b] into LDS; zero padding rows of s_f
    for (int i = tid; i < NN * DD; i += 256) s_x[i] = x_in[(size_t)b * NN * DD + i];
    for (int i = tid; i < (NP - NN) * DD; i += 256) s_f[NN * DD + i] = 0;
    __syncthreads();

    // ---------------- helper lambda-free LN builder done inline 3x ----------------
    // ======== graph learner: h0 = LN(x, layer0) (no bone) ========
    {
        const float* g = ln_g;  const float* bb = ln_b;
        for (int n = wave; n < NN; n += 8) {
            float vals[8]; float sum = 0.f, sq = 0.f;
#pragma unroll
            for (int e = 0; e < 8; ++e) {
                float v = s_x[n * DD + lane * 8 + e];
                vals[e] = v; sum += v; sq += v * v;
            }
            for (int m = 16; m > 0; m >>= 1) { sum += __shfl_xor(sum, m, 32); sq += __shfl_xor(sq, m, 32); }
            float mean = sum * (1.f / DD);
            float inv = rsqrtf(sq * (1.f / DD) - mean * mean + 1e-5f);
#pragma unroll
            for (int e = 0; e < 8; ++e) {
                int d = lane * 8 + e;
                s_f[n * DD + d] = f2bf((vals[e] - mean) * inv * g[d] + bb[d]);
            }
        }
    }
    __syncthreads();

    // xt = h0 @ gl_w  -> s_q (A-layout bf16)
    for (int t = wave; t < 32; t += 8) {
        int ct = t >> 1, mt = t & 1;
        v8f acc = {};
        for (int k = 0; k < DD; k += 32)
            acc = wmma_bf16(load_tileA(s_f, DD, mt * 16, k), load_tileB(glwT, DD, ct * 16, k), acc);
        storeD_bf16(s_q, DD, mt * 16, ct * 16, acc, nullptr);
    }
    __syncthreads();

    // dadj_raw = xt @ h0^T -> s_scores[0..1023]
    if (wave < 4) {
        int mt = wave >> 1, ct = wave & 1;
        v8f acc = {};
        for (int k = 0; k < DD; k += 32)
            acc = wmma_bf16(load_tileA(s_q, DD, mt * 16, k), load_tileB(s_f, DD, ct * 16, k), acc);
        storeD_f32(s_scores, 32, mt * 16, ct * 16, acc);
    }
    __syncthreads();

    // relu + top-8 threshold -> s_dadj
    if (tid < NN) {
        float v[NN], tmp[NN];
#pragma unroll
        for (int m = 0; m < NN; ++m) {
            float t = s_scores[tid * 32 + m];
            v[m] = t > 0.f ? t : 0.f;
            tmp[m] = v[m];
        }
        float kth = 0.f;
        for (int it = 0; it < 8; ++it) {
            int bi = 0; float bvv = tmp[0];
#pragma unroll
            for (int m = 1; m < NN; ++m) if (tmp[m] > bvv) { bvv = tmp[m]; bi = m; }
            kth = bvv; tmp[bi] = -1e30f;
        }
#pragma unroll
        for (int m = 0; m < NN; ++m) s_dadj[tid * 32 + m] = (v[m] >= kth) ? v[m] : 0.f;
    }
    __syncthreads();

    // ======== two attention layers ========
    for (int l = 0; l < 2; ++l) {
        // f = LN(x, layer l) + bone
        {
            const float* g = ln_g + l * DD;  const float* bb = ln_b + l * DD;
            const float* bonep = bone + (size_t)b * NN * DD;
            for (int n = wave; n < NN; n += 8) {
                float vals[8]; float sum = 0.f, sq = 0.f;
#pragma unroll
                for (int e = 0; e < 8; ++e) {
                    float v = s_x[n * DD + lane * 8 + e];
                    vals[e] = v; sum += v; sq += v * v;
                }
                for (int m = 16; m > 0; m >>= 1) { sum += __shfl_xor(sum, m, 32); sq += __shfl_xor(sq, m, 32); }
                float mean = sum * (1.f / DD);
                float inv = rsqrtf(sq * (1.f / DD) - mean * mean + 1e-5f);
#pragma unroll
                for (int e = 0; e < 8; ++e) {
                    int d = lane * 8 + e;
                    float h = (vals[e] - mean) * inv * g[d] + bb[d] + bonep[n * DD + d];
                    s_f[n * DD + d] = f2bf(h);
                }
            }
        }
        __syncthreads();

        // q,k,v projections (96 tiles, K loop of 8)
        for (int t = wave; t < 96; t += 8) {
            int which = t >> 5, sub = t & 31;
            int ct = sub >> 1, mt = sub & 1;
            const ushort_t* W = ((which == 0) ? wq_bf : (which == 1) ? wk_bf : wv_bf) + l * DD * DD;
            v8f acc = {};
            for (int k = 0; k < DD; k += 32)
                acc = wmma_bf16(load_tileA(s_f, DD, mt * 16, k), load_tileB(W, DD, ct * 16, k), acc);
            const float* bias = ((which == 0) ? bq : (which == 1) ? bk : bv) + l * DD;
            if (which == 2)      storeD_vT(s_vT, mt * 16, ct * 16, acc, bias);
            else if (which == 0) storeD_bf16(s_q, DD, mt * 16, ct * 16, acc, bias);
            else                 storeD_bf16(s_k, DD, mt * 16, ct * 16, acc, bias);
        }
        __syncthreads();

        // scores = q_h k_h^T (one WMMA per tile, K = HD = 32)
        for (int t = wave; t < 32; t += 8) {
            int h = t >> 2, mt = (t >> 1) & 1, ct = t & 1;
            v8f acc = {};
            acc = wmma_bf16(load_tileA(s_q, DD, mt * 16, h * HDIM),
                            load_tileB(s_k, DD, ct * 16, h * HDIM), acc);
            storeD_f32(s_scores + h * 1024, 32, mt * 16, ct * 16, acc);
        }
        __syncthreads();

        // softmax + static adj + dadj gating; finalA -> s_q (bf16), attn -> global
        {
            int h = tid >> 5, n = tid & 31;
            ushort_t* fa = s_q + h * 1024 + n * 32;
            if (n < NN) {
                float r[NN]; float mx = -1e30f;
#pragma unroll
                for (int m = 0; m < NN; ++m) {
                    r[m] = s_scores[h * 1024 + n * 32 + m] * scale;
                    mx = fmaxf(mx, r[m]);
                }
                float sum = 0.f;
#pragma unroll
                for (int m = 0; m < NN; ++m) { r[m] = __expf(r[m] - mx); sum += r[m]; }
                float inv = 1.f / sum;
                size_t abase = XSZ + (size_t)l * ASZ +
                               ((size_t)b * HH + h) * (NN * NN) + (size_t)n * NN;
#pragma unroll
                for (int m = 0; m < NN; ++m) {
                    float fin = (r[m] * inv + (((ADJ_MASK[n] >> m) & 1u) ? 1.f : 0.f)) *
                                s_dadj[n * 32 + m];
                    out[abase + m] = fin;
                    fa[m] = f2bf(fin);
                }
#pragma unroll
                for (int m = NN; m < 32; ++m) fa[m] = 0;
            } else {
#pragma unroll
                for (int m = 0; m < 32; ++m) fa[m] = 0;
            }
        }
        __syncthreads();

        // out_h = final_adj_h @ v_h  -> s_out (aliases s_scores) as bf16 row-major
        ushort_t* s_out = (ushort_t*)s_scores;
        for (int t = wave; t < 32; t += 8) {
            int h = t >> 2, mt = (t >> 1) & 1, ct = t & 1;
            v8f acc = {};
            acc = wmma_bf16(load_tileA(s_q + h * 1024, 32, mt * 16, 0),
                            load_tileB(s_vT, 32, h * HDIM + ct * 16, 0), acc);
            storeD_bf16(s_out, DD, mt * 16, h * HDIM + ct * 16, acc, nullptr);
        }
        __syncthreads();

        // o-projection + residual into s_x
        for (int t = wave; t < 32; t += 8) {
            int ct = t >> 1, mt = t & 1;
            v8f acc = {};
            for (int k = 0; k < DD; k += 32)
                acc = wmma_bf16(load_tileA(s_out, DD, mt * 16, k),
                                load_tileB(wo_bf + l * DD * DD, DD, ct * 16, k), acc);
            int col = ct * 16 + (lane & 15);
            int mb = mt * 16 + ((lane >> 4) ? 8 : 0);
            float bias = bo[l * DD + col];
#pragma unroll
            for (int r = 0; r < 8; ++r) {
                int n = mb + r;
                if (n < NN) s_x[n * DD + col] += acc[r] + bias;
            }
        }
        __syncthreads();
    }

    // write final x
    for (int i = tid; i < NN * DD; i += 256) out[(size_t)b * NN * DD + i] = s_x[i];
}

extern "C" void kernel_launch(void* const* d_in, const int* in_sizes, int n_in,
                              void* d_out, int out_size, void* d_ws, size_t ws_size,
                              hipStream_t stream) {
    const float* x    = (const float*)d_in[0];
    const float* bone = (const float*)d_in[1];
    const float* ln_g = (const float*)d_in[2];
    const float* ln_b = (const float*)d_in[3];
    const float* wq   = (const float*)d_in[4];
    const float* bq   = (const float*)d_in[5];
    const float* wk   = (const float*)d_in[6];
    const float* bk   = (const float*)d_in[7];
    const float* wv   = (const float*)d_in[8];
    const float* bv   = (const float*)d_in[9];
    const float* wo   = (const float*)d_in[10];
    const float* bo   = (const float*)d_in[11];
    const float* glw  = (const float*)d_in[12];

    ushort_t* wbf = (ushort_t*)d_ws;   // 4*WSZ + D*D bf16 elems = 1.18 MB

    int prep_elems = 4 * WSZ + DD * DD;            // 589824
    prep_weights_kernel<<<(prep_elems + 255) / 256, 256, 0, stream>>>(wq, wk, wv, wo, glw, wbf);
    gnn_block_kernel<<<NB, 256, 0, stream>>>(x, bone, ln_g, ln_b, bq, bk, bv, bo, wbf,
                                             (float*)d_out);
}
